// GraphEncoder_89893665505774
// MI455X (gfx1250) — compile-verified, matrix operand order checked by reading, not run on hardware
//
#include <hip/hip_runtime.h>
#include <stddef.h>
#include <stdint.h>

// ---------------- problem constants (match reference) ----------------
#define N_NODES  50000
#define N_EDGES  1200000
#define N_REL    40
#define N_BASES  30
#define EMB      64
#define HID      128
#define N_GRAPHS 64

#define NPAD     50048                    // N rounded up to 128-row GEMM blocks
#define K1       (N_BASES*EMB + EMB)      // 1984  (basis-space + root concat)
#define K2       (N_BASES*HID + HID)      // 3968

typedef __attribute__((ext_vector_type(16))) __bf16 v16bf;
typedef __attribute__((ext_vector_type(8)))  __bf16 v8bf;
typedef __attribute__((ext_vector_type(8)))  float  v8f;

union V16U { v16bf v; v8bf h[2]; };

// ---------------- utility: zero bytes (grid-stride, n % 16 == 0) ----------------
__global__ void zero_bytes(void* p, size_t nbytes) {
  size_t i = (size_t)blockIdx.x * blockDim.x + threadIdx.x;
  size_t stride = (size_t)gridDim.x * blockDim.x;
  uint4* q = (uint4*)p;
  size_t m = nbytes >> 4;
  uint4 z; z.x = 0u; z.y = 0u; z.z = 0u; z.w = 0u;
  for (; i < m; i += stride) q[i] = z;
}

// ---------------- h0 = node_emb[x] -> bf16 into A1 root-tail ----------------
__global__ void gather_h0(const int* __restrict__ x, const float* __restrict__ emb,
                          __bf16* __restrict__ A1) {
  int tid = blockIdx.x * blockDim.x + threadIdx.x;
  if (tid >= N_NODES * EMB) return;
  int n = tid >> 6, i = tid & 63;
  float v = emb[(size_t)x[n] * EMB + i];
  A1[(size_t)n * K1 + N_BASES * EMB + i] = (__bf16)v;
}

// ---------------- pack weights column-major bf16: Wt[o][k] = W[k][o] ----------------
__global__ void prep_w(const float* __restrict__ basis, const float* __restrict__ root,
                       __bf16* __restrict__ Wt, int K, int Kbasis) {
  int tid = blockIdx.x * blockDim.x + threadIdx.x;
  if (tid >= HID * K) return;
  int o = tid / K, k = tid % K;
  float v = (k < Kbasis) ? basis[(size_t)k * HID + o]
                         : root[(size_t)(k - Kbasis) * HID + o];
  Wt[(size_t)o * K + k] = (__bf16)v;
}

// ---------------- per-(node,rel) edge counts ----------------
__global__ void count_edges(const int* __restrict__ ei, const int* __restrict__ et,
                            float* __restrict__ cnt) {
  int e = blockIdx.x * blockDim.x + threadIdx.x;
  if (e >= N_EDGES) return;
  int dst = ei[N_EDGES + e];
  int t = et[e];
  atomicAdd(&cnt[(size_t)dst * N_REL + t], 1.0f);
}

__global__ void count_graphs(const int* __restrict__ batch, int* __restrict__ gcnt) {
  int n = blockIdx.x * blockDim.x + threadIdx.x;
  if (n >= N_NODES) return;
  atomicAdd(&gcnt[batch[n]], 1);
}

// ---------------- device-scope packed bf16x2 atomic add (CDNA5) ----------------
__device__ __forceinline__ void atomic_pk_add_bf16(__bf16* addr, uint32_t pk) {
  asm volatile("global_atomic_pk_add_bf16 %0, %1, off scope:SCOPE_DEV"
               :: "v"((uint64_t)(uintptr_t)addr), "v"(pk) : "memory");
}

// ---------------- segment-sum of bf16 source features into bf16 agg ----------------
// messages gathered from the bf16 root-tail columns of the A matrix (stride = K, off = tail)
template<int C>
__global__ void scatter_msgs(const int* __restrict__ ei, const int* __restrict__ et,
                             const __bf16* __restrict__ h, int hStride, int hOff,
                             __bf16* __restrict__ agg) {
  long long tid = (long long)blockIdx.x * blockDim.x + threadIdx.x;
  long long total = (long long)N_EDGES * (C / 2);
  if (tid >= total) return;
  int e  = (int)(tid / (C / 2));
  int i2 = 2 * (int)(tid % (C / 2));
  int src = ei[e];
  int dst = ei[N_EDGES + e];
  int t = et[e];
  uint32_t pk = *(const uint32_t*)(h + (size_t)src * hStride + hOff + i2);
  atomic_pk_add_bf16(agg + ((size_t)dst * N_REL + t) * C + i2, pk);
}

// ---------------- mean-normalize + contract rel->basis, emit bf16 A matrix ----------------
// A[n, b*C + i] = sum_r comp[r,b] * agg[n,r,i] / max(cnt[n,r],1)
template<int C>
__global__ void contract_rb(const __bf16* __restrict__ agg, const float* __restrict__ cnt,
                            const float* __restrict__ comp, __bf16* __restrict__ A, int Kout) {
  __shared__ float sComp[N_REL * N_BASES];
  __shared__ float sScale[N_REL];
  int n = blockIdx.x;
  int i = threadIdx.x;                  // C threads per block (one node per block)
  for (int j = i; j < N_REL * N_BASES; j += C) sComp[j] = comp[j];
  if (i < N_REL) {
    float c = cnt[(size_t)n * N_REL + i];
    sScale[i] = 1.0f / fmaxf(c, 1.0f);
  }
  __syncthreads();
  float v[N_REL];
#pragma unroll
  for (int r = 0; r < N_REL; ++r)
    v[r] = (float)agg[((size_t)n * N_REL + r) * C + i] * sScale[r];
  for (int b = 0; b < N_BASES; ++b) {
    float s = 0.0f;
#pragma unroll
    for (int r = 0; r < N_REL; ++r) s += v[r] * sComp[r * N_BASES + b];
    A[(size_t)n * Kout + b * C + i] = (__bf16)s;
  }
}

// ---------------- WMMA mainloop: one wave computes a 16x128 f32 strip ----------------
// A fragment (bf16 16x32): lane L holds row m=L&15; two contiguous 8-elem chunks at
// k0+8*(L>=16) and k0+16+8*(L>=16). B stored column-major -> identical pattern.
__device__ __forceinline__ void wmma_strip(const __bf16* __restrict__ A,
                                           const __bf16* __restrict__ Wt,
                                           int K, int row0, int lane, v8f acc[8]) {
  const int m  = lane & 15;
  const int kb = (lane >> 4) << 3;      // 0 or 8
  const __bf16* aRow = A + (size_t)(row0 + m) * K;
  for (int k0 = 0; k0 < K; k0 += 32) {
    V16U a;
    a.h[0] = *(const v8bf*)(aRow + k0 + kb);
    a.h[1] = *(const v8bf*)(aRow + k0 + 16 + kb);
#pragma unroll
    for (int t = 0; t < 8; ++t) {
      const __bf16* bCol = Wt + (size_t)(t * 16 + m) * K + k0 + kb;
      V16U b;
      b.h[0] = *(const v8bf*)(bCol);
      b.h[1] = *(const v8bf*)(bCol + 16);
      acc[t] = __builtin_amdgcn_wmma_f32_16x16x32_bf16(
          false, a.v, false, b.v, (short)0, acc[t], false, false);
    }
  }
}

// ---------------- GEMM layer1: relu(A1 @ W1 + bias) -> bf16 into A2 root-tail ----------
__global__ void gemm_layer1(const __bf16* __restrict__ A, const __bf16* __restrict__ Wt,
                            const float* __restrict__ bias, __bf16* __restrict__ A2) {
  int wave = threadIdx.x >> 5, lane = threadIdx.x & 31;
  int row0 = (blockIdx.x * 8 + wave) * 16;
  v8f zero = {0.f, 0.f, 0.f, 0.f, 0.f, 0.f, 0.f, 0.f};
  v8f acc[8];
#pragma unroll
  for (int t = 0; t < 8; ++t) acc[t] = zero;
  wmma_strip(A, Wt, K1, row0, lane, acc);
  // C/D layout: VGPR i of lane L -> (row = row0 + i + 8*(L>=16), col = t*16 + (L&15))
  int colLane = lane & 15;
  int rowHalf = (lane >> 4) << 3;
#pragma unroll
  for (int t = 0; t < 8; ++t) {
    int col = t * 16 + colLane;
    float bc = bias[col];
#pragma unroll
    for (int i = 0; i < 8; ++i) {
      int row = row0 + rowHalf + i;
      if (row < N_NODES) {
        float v = fmaxf(acc[t][i] + bc, 0.0f);
        A2[(size_t)row * K2 + N_BASES * HID + col] = (__bf16)v;   // h1 (root-concat tail)
      }
    }
  }
}

// ---------------- GEMM layer2 + fused global mean-pool (sum part) ----------------
__global__ void gemm_layer2(const __bf16* __restrict__ A, const __bf16* __restrict__ Wt,
                            const float* __restrict__ bias, const int* __restrict__ batch,
                            float* __restrict__ out) {
  int wave = threadIdx.x >> 5, lane = threadIdx.x & 31;
  int row0 = (blockIdx.x * 8 + wave) * 16;
  v8f zero = {0.f, 0.f, 0.f, 0.f, 0.f, 0.f, 0.f, 0.f};
  v8f acc[8];
#pragma unroll
  for (int t = 0; t < 8; ++t) acc[t] = zero;
  wmma_strip(A, Wt, K2, row0, lane, acc);
  int colLane = lane & 15;
  int rowHalf = (lane >> 4) << 3;
#pragma unroll
  for (int t = 0; t < 8; ++t) {
    int col = t * 16 + colLane;
    float bc = bias[col];
#pragma unroll
    for (int i = 0; i < 8; ++i) {
      int row = row0 + rowHalf + i;
      if (row < N_NODES) {
        float v = fmaxf(acc[t][i] + bc, 0.0f);
        int g = batch[row];
        atomicAdd(&out[(size_t)g * HID + col], v);
      }
    }
  }
}

__global__ void finalize_pool(float* __restrict__ out, const int* __restrict__ gcnt) {
  int tid = blockIdx.x * blockDim.x + threadIdx.x;
  if (tid >= N_GRAPHS * HID) return;
  int g = tid / HID;
  out[tid] /= fmaxf((float)gcnt[g], 1.0f);
}

// ---------------- launcher ----------------
extern "C" void kernel_launch(void* const* d_in, const int* in_sizes, int n_in,
                              void* d_out, int out_size, void* d_ws, size_t ws_size,
                              hipStream_t stream) {
  const int*   x      = (const int*)d_in[0];
  const int*   ei     = (const int*)d_in[1];   // [2, E] flat
  const int*   et     = (const int*)d_in[2];
  const int*   batch  = (const int*)d_in[3];
  const float* emb    = (const float*)d_in[4];
  const float* comp1  = (const float*)d_in[5];
  const float* basis1 = (const float*)d_in[6];
  const float* root1  = (const float*)d_in[7];
  const float* bias1  = (const float*)d_in[8];
  const float* comp2  = (const float*)d_in[9];
  const float* basis2 = (const float*)d_in[10];
  const float* root2  = (const float*)d_in[11];
  const float* bias2  = (const float*)d_in[12];
  float* out = (float*)d_out;

  // bump allocator over d_ws (hipMalloc base is 256B aligned)
  char* ws = (char*)d_ws;
  size_t off = 0;
  auto alloc = [&](size_t bytes) -> void* {
    void* p = ws + off;
    off = (off + bytes + 255) & ~(size_t)255;
    return p;
  };
  float*  cnt  = (float*) alloc((size_t)N_NODES * N_REL * 4);
  int*    gcnt = (int*)   alloc(256);
  __bf16* agg  = (__bf16*)alloc((size_t)N_NODES * N_REL * HID * 2);   // reused both layers
  __bf16* A1   = (__bf16*)alloc((size_t)NPAD * K1 * 2);
  __bf16* A2   = (__bf16*)alloc((size_t)NPAD * K2 * 2);
  __bf16* Wt1  = (__bf16*)alloc((size_t)HID * K1 * 2);
  __bf16* Wt2  = (__bf16*)alloc((size_t)HID * K2 * 2);

  const int B = 256;
  const size_t aggBytes1 = (size_t)N_NODES * N_REL * EMB * 2;   // layer-1 region only
  const size_t aggBytes2 = (size_t)N_NODES * N_REL * HID * 2;

  // ---- init (ws + out are poisoned by harness) ----
  zero_bytes<<<2048, B, 0, stream>>>(cnt,  (size_t)N_NODES * N_REL * 4);
  zero_bytes<<<1,    64, 0, stream>>>(gcnt, 256);
  zero_bytes<<<4096, B, 0, stream>>>(agg,  aggBytes1);
  zero_bytes<<<64,   B, 0, stream>>>(out,  (size_t)N_GRAPHS * HID * 4);
  zero_bytes<<<64,   B, 0, stream>>>(A1 + (size_t)N_NODES * K1, (size_t)(NPAD - N_NODES) * K1 * 2);
  zero_bytes<<<64,   B, 0, stream>>>(A2 + (size_t)N_NODES * K2, (size_t)(NPAD - N_NODES) * K2 * 2);

  // ---- weight packing + h0 gather (bf16 into A1 tail) ----
  prep_w<<<(HID * K1 + B - 1) / B, B, 0, stream>>>(basis1, root1, Wt1, K1, N_BASES * EMB);
  prep_w<<<(HID * K2 + B - 1) / B, B, 0, stream>>>(basis2, root2, Wt2, K2, N_BASES * HID);
  gather_h0<<<(N_NODES * EMB + B - 1) / B, B, 0, stream>>>(x, emb, A1);

  // ---- degree counts (shared by both layers) + graph counts ----
  count_edges<<<(N_EDGES + B - 1) / B, B, 0, stream>>>(ei, et, cnt);
  count_graphs<<<(N_NODES + B - 1) / B, B, 0, stream>>>(batch, gcnt);

  // ---- layer 1: messages are bf16 h0 living in A1's root tail ----
  {
    long long tot = (long long)N_EDGES * (EMB / 2);
    scatter_msgs<EMB><<<(unsigned)((tot + B - 1) / B), B, 0, stream>>>(
        ei, et, A1, K1, N_BASES * EMB, agg);
    contract_rb<EMB><<<N_NODES, EMB, 0, stream>>>(agg, cnt, comp1, A1, K1);
    gemm_layer1<<<NPAD / 128, 256, 0, stream>>>(A1, Wt1, bias1, A2);
  }

  // ---- layer 2: messages are bf16 h1 living in A2's root tail ----
  {
    zero_bytes<<<4096, B, 0, stream>>>(agg, aggBytes2);   // re-zero after layer-1 use
    long long tot = (long long)N_EDGES * (HID / 2);
    scatter_msgs<HID><<<(unsigned)((tot + B - 1) / B), B, 0, stream>>>(
        ei, et, A2, K2, N_BASES * HID, agg);
    contract_rb<HID><<<N_NODES, HID, 0, stream>>>(agg, cnt, comp2, A2, K2);
    gemm_layer2<<<NPAD / 128, 256, 0, stream>>>(A2, Wt2, bias2, batch, out);
  }

  // ---- mean pool divide ----
  finalize_pool<<<(N_GRAPHS * HID + B - 1) / B, B, 0, stream>>>(out, gcnt);

  (void)in_sizes; (void)n_in; (void)out_size; (void)ws_size;
}
